// WordGenerator_65678639890947
// MI455X (gfx1250) — compile-verified, matrix operand order checked by reading, not run on hardware
//
#include <hip/hip_runtime.h>
#include <hip/hip_bf16.h>
#include <cstdint>

#define BB  256
#define TT  128
#define HD1 256
#define HD2 512
#define VV  50257

typedef __attribute__((ext_vector_type(16))) _Float16     v16h;
typedef __attribute__((ext_vector_type(8)))  float        v8f;
typedef __attribute__((ext_vector_type(2)))  float        v2f;
typedef __attribute__((ext_vector_type(4)))  unsigned int v4u;

union Frag16 { v16h h; v4u q[2]; };

__device__ __forceinline__ float sigf(float x) { return 1.0f / (1.0f + expf(-x)); }

// ---------------------------------------------------------------- converts
__global__ void cvt_f32_to_f16(const float* __restrict__ s, _Float16* __restrict__ d, int n) {
  for (int i = blockIdx.x * blockDim.x + threadIdx.x; i < n; i += gridDim.x * blockDim.x)
    d[i] = (_Float16)s[i];
}

// seq[p*H + h] = emb[x[p]*H + h]   (p enumerates (t,b) of the reshape — a pure reshape)
__global__ void embed_gather(const int* __restrict__ x, const float* __restrict__ emb,
                             _Float16* __restrict__ seq, int n) {
  for (int i = blockIdx.x * blockDim.x + threadIdx.x; i < n; i += gridDim.x * blockDim.x) {
    int p = i >> 8;          // / HD1
    int h = i & (HD1 - 1);
    int tok = x[p];
    seq[i] = (_Float16)emb[(size_t)tok * HD1 + h];
  }
}

// -------------------------------------------------- WMMA fragment loading
// CDNA5 16-bit A 16x32 layout: lane l holds row m=l&15; K halves split by l>>4.
// Per lane the fragment is two contiguous 16B runs -> two global_load_b128.
// B (32x16, K-major from W rows) uses the mirrored mapping with n=l&15.
__device__ __forceinline__ void load_frag(const _Float16* __restrict__ base, int ld,
                                          int row, int k0, int lane, Frag16& f) {
  const _Float16* p = base + (size_t)row * ld + k0 + ((lane >> 4) * 8);
  f.q[0] = *(const v4u*)(p);        // v0..v3 : k-pairs {0,2,4,6}(+1)
  f.q[1] = *(const v4u*)(p + 16);   // v4..v7 : k-pairs {16,18,20,22}(+1)
}

// ---------------------------------------------------------- fused LSTM step
// Block = 4 waves; wave g computes gate g (i,f,g,o) 16x16 tile with WMMA chain,
// LDS exchange, then elementwise c/h update.
template <int KIN, int KHID, int HG>
__global__ __launch_bounds__(128) void lstm_step(
    const _Float16* __restrict__ xin, int ldx,
    const _Float16* __restrict__ hprev, int ldh,
    const _Float16* __restrict__ Wih,      // (4*HG, KIN)  f16 row-major
    const _Float16* __restrict__ Whh,      // (4*HG, KHID) f16 row-major
    const float* __restrict__ bih, const float* __restrict__ bhh,
    const float* __restrict__ cprev, int ldcp,
    float* __restrict__ cout, int ldco,
    _Float16* __restrict__ hout, int ldho,
    float* __restrict__ hout32) {
  const int wave = threadIdx.x >> 5;
  const int lane = threadIdx.x & 31;
  const int brow = blockIdx.x * 16 + (lane & 15);
  const int wrow = wave * HG + blockIdx.y * 16 + (lane & 15);

  v8f acc = {};
#pragma unroll 4
  for (int k0 = 0; k0 < KIN; k0 += 32) {
    Frag16 a, b;
    load_frag(xin, ldx, brow, k0, lane, a);
    load_frag(Wih, KIN, wrow, k0, lane, b);
    acc = __builtin_amdgcn_wmma_f32_16x16x32_f16(false, a.h, false, b.h,
                                                 (short)0, acc, false, false);
  }
#pragma unroll 4
  for (int k0 = 0; k0 < KHID; k0 += 32) {
    Frag16 a, b;
    load_frag(hprev, ldh, brow, k0, lane, a);
    load_frag(Whh, KHID, wrow, k0, lane, b);
    acc = __builtin_amdgcn_wmma_f32_16x16x32_f16(false, a.h, false, b.h,
                                                 (short)0, acc, false, false);
  }

  __shared__ float gl[4][16][16];
  {
    const int mh = (lane >> 4) * 8;
#pragma unroll
    for (int v = 0; v < 8; ++v) gl[wave][v + mh][lane & 15] = acc[v];
  }
  __syncthreads();

  for (int e = threadIdx.x; e < 256; e += 128) {
    const int mm = e >> 4, nn = e & 15;
    const int gb  = blockIdx.y * 16 + nn;
    const int row = blockIdx.x * 16 + mm;
    float iv = gl[0][mm][nn] + bih[0 * HG + gb] + bhh[0 * HG + gb];
    float fv = gl[1][mm][nn] + bih[1 * HG + gb] + bhh[1 * HG + gb];
    float gv = gl[2][mm][nn] + bih[2 * HG + gb] + bhh[2 * HG + gb];
    float ov = gl[3][mm][nn] + bih[3 * HG + gb] + bhh[3 * HG + gb];
    float cp = cprev[(size_t)row * ldcp + gb];
    float c2 = sigf(fv) * cp + sigf(iv) * tanhf(gv);
    float h2 = sigf(ov) * tanhf(c2);
    cout[(size_t)row * ldco + gb] = c2;
    hout[(size_t)row * ldho + gb] = (_Float16)h2;
    if (hout32) hout32[(size_t)row * HG + gb] = h2;
  }
}

// ---------------------------------------------------- final vocab projection
// out[b][v] = h3[b] . Wout[v] + bout[v], f32 WMMA 16x16x4 (Wout stays fp32; L2-resident)
__global__ __launch_bounds__(128) void proj_f32(
    const float* __restrict__ h3, const float* __restrict__ Wout,
    const float* __restrict__ bout, float* __restrict__ out) {
  const int NT = (VV + 15) / 16;
  const int wave = threadIdx.x >> 5, lane = threadIdx.x & 31;
  const int nt = blockIdx.y * 4 + wave;
  if (nt >= NT) return;
  const int n  = lane & 15;
  int vc = nt * 16 + n; if (vc > VV - 1) vc = VV - 1;
  const int r  = blockIdx.x * 16 + n;
  const int kl = (lane >> 4) * 2;

  v8f acc = {};
#if __has_builtin(__builtin_amdgcn_wmma_f32_16x16x4_f32)
  for (int k0 = 0; k0 < HD2; k0 += 4) {
    __builtin_prefetch(&Wout[(size_t)vc * HD2 + k0 + 64], 0, 1);  // speculative, safe if OOB
    v2f a, b;
    a.x = h3[(size_t)r * HD2 + k0 + kl];
    a.y = h3[(size_t)r * HD2 + k0 + kl + 1];
    b.x = Wout[(size_t)vc * HD2 + k0 + kl];
    b.y = Wout[(size_t)vc * HD2 + k0 + kl + 1];
    acc = __builtin_amdgcn_wmma_f32_16x16x4_f32(false, a, false, b,
                                                (short)0, acc, false, false);
  }
#else
#pragma unroll
  for (int v = 0; v < 8; ++v) {
    int row = blockIdx.x * 16 + v + (lane >> 4) * 8;
    float s = 0.f;
    for (int k = 0; k < HD2; ++k) s += h3[(size_t)row * HD2 + k] * Wout[(size_t)vc * HD2 + k];
    acc[v] = s;
  }
#endif

#pragma unroll
  for (int v = 0; v < 8; ++v) {
    const int row = blockIdx.x * 16 + v + (lane >> 4) * 8;
    const int col = nt * 16 + (lane & 15);
    if (col < VV) out[(size_t)row * VV + col] = acc[v] + bout[col];
  }
}

// ================================================================== host
extern "C" void kernel_launch(void* const* d_in, const int* in_sizes, int n_in,
                              void* d_out, int out_size, void* d_ws, size_t ws_size,
                              hipStream_t stream) {
  (void)in_sizes; (void)n_in; (void)out_size; (void)ws_size;
  char* ws = (char*)d_ws;
  size_t off = 0;
  auto alloc = [&](size_t bytes) -> char* {
    char* p = ws + off;
    off = (off + bytes + 255) & ~(size_t)255;
    return p;
  };

  const int*   x     = (const int*)d_in[0];
  const float* emb   = (const float*)d_in[1];
  const float* bih_f = (const float*)d_in[4],  *bhh_f = (const float*)d_in[5];
  const float* bih_b = (const float*)d_in[8],  *bhh_b = (const float*)d_in[9];
  const float* b1ih  = (const float*)d_in[12], *b1hh  = (const float*)d_in[13];
  const float* b2ih  = (const float*)d_in[16], *b2hh  = (const float*)d_in[17];
  const float* b3ih  = (const float*)d_in[20], *b3hh  = (const float*)d_in[21];
  const float* Wout  = (const float*)d_in[22], *bout  = (const float*)d_in[23];
  const float* c0f = (const float*)d_in[25], *c0b = (const float*)d_in[27];
  const float* c01 = (const float*)d_in[29], *c02 = (const float*)d_in[31];
  const float* c03 = (const float*)d_in[33];

  // workspace
  _Float16* seq   = (_Float16*)alloc((size_t)TT * BB * HD1 * 2);
  _Float16* pairs = (_Float16*)alloc((size_t)TT * BB * HD2 * 2);
  const size_t wsm = (size_t)4 * HD1 * HD1 * 2;   // small weight (1024x256)
  const size_t wbg = (size_t)4 * HD2 * HD2 * 2;   // big weight   (2048x512)
  _Float16* wihF = (_Float16*)alloc(wsm); _Float16* whhF = (_Float16*)alloc(wsm);
  _Float16* wihB = (_Float16*)alloc(wsm); _Float16* whhB = (_Float16*)alloc(wsm);
  _Float16* w1i = (_Float16*)alloc(wbg); _Float16* w1h = (_Float16*)alloc(wbg);
  _Float16* w2i = (_Float16*)alloc(wbg); _Float16* w2h = (_Float16*)alloc(wbg);
  _Float16* w3i = (_Float16*)alloc(wbg); _Float16* w3h = (_Float16*)alloc(wbg);
  _Float16* h0F = (_Float16*)alloc((size_t)BB * HD1 * 2);
  _Float16* h0B = (_Float16*)alloc((size_t)BB * HD1 * 2);
  _Float16* h0s[3]; float* cbufS[3][2]; _Float16* hbufS[3][2];
  for (int i = 0; i < 3; ++i) {
    h0s[i] = (_Float16*)alloc((size_t)BB * HD2 * 2);
    for (int j = 0; j < 2; ++j) {
      cbufS[i][j] = (float*)alloc((size_t)BB * HD2 * 4);
      hbufS[i][j] = (_Float16*)alloc((size_t)BB * HD2 * 2);
    }
  }
  float* cF[2] = {(float*)alloc((size_t)BB * HD1 * 4), (float*)alloc((size_t)BB * HD1 * 4)};
  float* cB[2] = {(float*)alloc((size_t)BB * HD1 * 4), (float*)alloc((size_t)BB * HD1 * 4)};
  float* h3f32 = (float*)alloc((size_t)BB * HD2 * 4);

  auto cvt = [&](int idx, _Float16* dst, int n) {
    cvt_f32_to_f16<<<dim3(256), dim3(256), 0, stream>>>((const float*)d_in[idx], dst, n);
  };
  const int nsm = 4 * HD1 * HD1, nbg = 4 * HD2 * HD2;
  cvt(2, wihF, nsm);  cvt(3, whhF, nsm);
  cvt(6, wihB, nsm);  cvt(7, whhB, nsm);
  cvt(10, w1i, nbg);  cvt(11, w1h, nbg);
  cvt(14, w2i, nbg);  cvt(15, w2h, nbg);
  cvt(18, w3i, nbg);  cvt(19, w3h, nbg);
  cvt(24, h0F, BB * HD1); cvt(26, h0B, BB * HD1);
  cvt(28, h0s[0], BB * HD2); cvt(30, h0s[1], BB * HD2); cvt(32, h0s[2], BB * HD2);

  embed_gather<<<dim3(4096), dim3(256), 0, stream>>>(x, emb, seq, TT * BB * HD1);

  const dim3 gridS(BB / 16, HD1 / 16), gridB(BB / 16, HD2 / 16), blk(128);

  // bidirectional scan: fwd h_t -> pairs[t][0:H], bwd h of step t (input seq[TT-1-t]) -> pairs[t][H:2H]
  for (int t = 0; t < TT; ++t) {
    const _Float16* hpF = (t == 0) ? h0F : pairs + (size_t)(t - 1) * BB * HD2;
    const int ldhF = (t == 0) ? HD1 : HD2;
    const float* cpF = (t == 0) ? c0f : cF[(t + 1) & 1];
    lstm_step<HD1, HD1, HD1><<<gridS, blk, 0, stream>>>(
        seq + (size_t)t * BB * HD1, HD1, hpF, ldhF, wihF, whhF, bih_f, bhh_f,
        cpF, HD1, cF[t & 1], HD1, pairs + (size_t)t * BB * HD2, HD2, nullptr);

    const _Float16* hpB = (t == 0) ? h0B : pairs + (size_t)(t - 1) * BB * HD2 + HD1;
    const int ldhB = (t == 0) ? HD1 : HD2;
    const float* cpB = (t == 0) ? c0b : cB[(t + 1) & 1];
    lstm_step<HD1, HD1, HD1><<<gridS, blk, 0, stream>>>(
        seq + (size_t)(TT - 1 - t) * BB * HD1, HD1, hpB, ldhB, wihB, whhB, bih_b, bhh_b,
        cpB, HD1, cB[t & 1], HD1, pairs + (size_t)t * BB * HD2 + HD1, HD2, nullptr);
  }

  // 3 stacked cells
  const _Float16* wI[3] = {w1i, w2i, w3i};
  const _Float16* wH[3] = {w1h, w2h, w3h};
  const float* bI[3] = {b1ih, b2ih, b3ih};
  const float* bH[3] = {b1hh, b2hh, b3hh};
  const float* c0s[3] = {c01, c02, c03};
  for (int t = 0; t < TT; ++t) {
    for (int L = 0; L < 3; ++L) {
      const _Float16* xin = (L == 0) ? pairs + (size_t)t * BB * HD2 : hbufS[L - 1][t & 1];
      const _Float16* hp  = (t == 0) ? h0s[L] : hbufS[L][(t + 1) & 1];
      const float*    cp  = (t == 0) ? c0s[L] : cbufS[L][(t + 1) & 1];
      float* h32 = (L == 2 && t == TT - 1) ? h3f32 : nullptr;
      lstm_step<HD2, HD2, HD2><<<gridB, blk, 0, stream>>>(
          xin, HD2, hp, HD2, wI[L], wH[L], bI[L], bH[L],
          cp, HD2, cbufS[L][t & 1], HD2, hbufS[L][t & 1], HD2, h32);
    }
  }

  // projection
  const int NT = (VV + 15) / 16;
  proj_f32<<<dim3(BB / 16, (NT + 3) / 4), blk, 0, stream>>>(h3f32, Wout, bout, (float*)d_out);
}
